// Mamba_57586921504987
// MI455X (gfx1250) — matrix operand order repaired and checked by reading, not compiled, not run on hardware
//
#include <hip/hip_runtime.h>

// ---------------- problem constants ----------------
#define DM     2048           // d_model
#define DI     4096           // d_inner
#define LSEQ   2048           // sequence length
#define NBATCH 2
#define NTOK   (NBATCH * LSEQ)   // 4096 tokens
#define DSTATE 16
#define DTR    128            // dt_rank
#define EOUT   (2 * DI)       // 8192 rows of W_in
#define XDBL   (DTR + 2 * DSTATE) // 160

// ---------------- WMMA types ----------------
typedef __attribute__((ext_vector_type(16))) __bf16 v16bf;
typedef __attribute__((ext_vector_type(8)))  float  v8f;
typedef __attribute__((ext_vector_type(4)))  int    v4i;

// CDNA5 async global->LDS copy path (ASYNCcnt), if the toolchain exposes it.
#if defined(__AMDGCN__) && __has_builtin(__builtin_amdgcn_global_load_async_to_lds_b128)
#define USE_ASYNC_COPY 1
#else
#define USE_ASYNC_COPY 0
#endif

#if USE_ASYNC_COPY
typedef __attribute__((address_space(1))) v4i gv4i;   // global int4
typedef __attribute__((address_space(3))) v4i lv4i;   // LDS int4
__device__ __forceinline__ void async_b128(const void* g, void* l) {
  __builtin_amdgcn_global_load_async_to_lds_b128((gv4i*)(void*)g, (lv4i*)l, 0, 0);
}
__device__ __forceinline__ void wait_async0() {
#if __has_builtin(__builtin_amdgcn_s_wait_asynccnt)
  __builtin_amdgcn_s_wait_asynccnt(0);
#else
  asm volatile("s_wait_asynccnt 0x0" ::: "memory");
#endif
}
#endif

__device__ __forceinline__ __bf16 f2bf(float f) {
  unsigned u = __builtin_bit_cast(unsigned, f);
  unsigned r = (u + 0x7FFFu + ((u >> 16) & 1u)) >> 16;   // round-to-nearest-even
  return __builtin_bit_cast(__bf16, (unsigned short)r);
}

__device__ __forceinline__ float silu_f(float v) {
  return v / (1.0f + __expf(-v));
}

// ---------------- f32 -> bf16 conversion ----------------
__global__ void cvt_f32_bf16(const float* __restrict__ s, __bf16* __restrict__ d, long n) {
  long i = (long)blockIdx.x * blockDim.x + threadIdx.x;
  long stride = (long)gridDim.x * blockDim.x;
  for (; i < n; i += stride) d[i] = f2bf(s[i]);
}

// ---------------- bf16 WMMA GEMM: C[M,N] = A[M,K] * B[N,K]^T ----------------
// M, K multiples of TBM/TBK at all call sites. NGUARD: N % TBN != 0 (x_dbl GEMM only).
// MODE 0: plain f32 store to C0 (ldc = N)
// MODE 1: deinterleave columns: even -> C0, odd -> C1 (each M x N/2)
// MODE 2: split x_dbl: col<128 -> Cb (bf16); 128..143 -> C1; 144..159 -> C2
// MODE 3: softplus(acc + bias[col]) -> C0 (ldc = N)
#define TBM 128
#define TBN 128
#define TBK 32
#define LDS_STR 40                       // padded row stride (80B: 16B-aligned, conflict-free)
#define BUFSZ ((TBM + TBN) * LDS_STR)    // one double-buffer slice (bf16 elems)

template<int MODE, bool NGUARD>
__global__ __launch_bounds__(256)
void gemm_bf16_wmma(const __bf16* __restrict__ A, const __bf16* __restrict__ B,
                    int M, int N, int K,
                    float* __restrict__ C0, float* __restrict__ C1,
                    float* __restrict__ C2, __bf16* __restrict__ Cb,
                    const float* __restrict__ bias)
{
  __shared__ __bf16 smem[2 * BUFSZ];     // [buf][A(128x40) | B(128x40)]

  const int tid   = threadIdx.x;
  const int lane  = tid & 31;
  const int wave  = tid >> 5;            // 8 waves
  const int wm    = wave & 1;            // 2 x 64-row halves
  const int wn    = wave >> 1;           // 4 x 32-col strips
  const int lrow  = lane & 15;
  const int khalf = lane >> 4;

  const int m0 = blockIdx.y * TBM;
  const int n0 = blockIdx.x * TBN;

  // loader coordinates: 512 (row, 8-elem quarter) slots over 256 threads
  int ldRow[2], ldCol[2];
#pragma unroll
  for (int i = 0; i < 2; i++) {
    int lin = tid + i * 256;
    ldRow[i] = lin >> 2;
    ldCol[i] = (lin & 3) << 3;
  }

  v8f acc[4][2];
  const v8f zacc = {0.f,0.f,0.f,0.f,0.f,0.f,0.f,0.f};
#pragma unroll
  for (int mt = 0; mt < 4; mt++)
#pragma unroll
    for (int nt = 0; nt < 2; nt++) acc[mt][nt] = zacc;

  // B row index with clamp: OOB rows feed only accumulator columns the epilogue drops.
  auto browc = [&](int i) {
    int gc = n0 + ldRow[i];
    if (NGUARD && gc >= N) gc = N - 1;
    return gc;
  };

  // ---- fragment load + 8 WMMAs for one LDS buffer ----
  auto compute = [&](int sel) {
    const __bf16* curA = smem + sel * BUFSZ;
    const __bf16* curB = curA + TBM * LDS_STR;
    // A fragment (16x32): lane (khalf,m): elems 0-7 -> K khalf*8.., 8-15 -> K 16+khalf*8..
    v16bf afr[4], bfr[2];
#pragma unroll
    for (int mt = 0; mt < 4; mt++) {
      const __bf16* p = curA + (wm * 64 + mt * 16 + lrow) * LDS_STR + khalf * 8;
#pragma unroll
      for (int i = 0; i < 8; i++) { afr[mt][i] = p[i]; afr[mt][8 + i] = p[16 + i]; }
    }
    // B fragment (32x16): lane (khalf,n): elems 0-7 -> K khalf*8.., 8-15 -> K 16+khalf*8..
#pragma unroll
    for (int nt = 0; nt < 2; nt++) {
      const __bf16* p = curB + (wn * 32 + nt * 16 + lrow) * LDS_STR + khalf * 8;
#pragma unroll
      for (int i = 0; i < 8; i++) { bfr[nt][i] = p[i]; bfr[nt][8 + i] = p[16 + i]; }
    }
#pragma unroll
    for (int mt = 0; mt < 4; mt++)
#pragma unroll
      for (int nt = 0; nt < 2; nt++)
        acc[mt][nt] = __builtin_amdgcn_wmma_f32_16x16x32_bf16(
            false, afr[mt], false, bfr[nt], (short)0, acc[mt][nt], false, false);
  };

  const int KT = K / TBK;

#if USE_ASYNC_COPY
  // ---- async copy engine stages tiles; zero staging VGPRs, ASYNCcnt pipelining ----
  auto async_tile = [&](int sel, int k0) {
    __bf16* dA = smem + sel * BUFSZ;
    __bf16* dB = dA + TBM * LDS_STR;
#pragma unroll
    for (int i = 0; i < 2; i++) {
      async_b128(A + (size_t)(m0 + ldRow[i]) * K + k0 + ldCol[i],
                 dA + ldRow[i] * LDS_STR + ldCol[i]);
      async_b128(B + (size_t)browc(i) * K + k0 + ldCol[i],
                 dB + ldRow[i] * LDS_STR + ldCol[i]);
    }
  };
  async_tile(0, 0);
  for (int kt = 0; kt < KT; kt++) {
    wait_async0();                         // my copies for tile kt have landed in LDS
    __syncthreads();                       // everyone's landed; buf[(kt+1)&1] free to overwrite
    if (kt + 1 < KT) async_tile((kt + 1) & 1, (kt + 1) * TBK);  // overlaps compute below
    compute(kt & 1);
  }
#else
  // ---- fallback: register staging, peeled loop (no branches across staged regs) ----
  uint4 ra[2], rb[2];
  auto stage = [&](int k0) {
#pragma unroll
    for (int i = 0; i < 2; i++) {
      ra[i] = *(const uint4*)(A + (size_t)(m0 + ldRow[i]) * K + k0 + ldCol[i]);
      rb[i] = *(const uint4*)(B + (size_t)browc(i) * K + k0 + ldCol[i]);
      __builtin_prefetch(A + (size_t)(m0 + ldRow[i]) * K + k0 + TBK + ldCol[i], 0, 0);
      __builtin_prefetch(B + (size_t)browc(i) * K + k0 + TBK + ldCol[i], 0, 0);
    }
  };
  auto commit = [&](int sel) {
    __bf16* dA = smem + sel * BUFSZ;
    __bf16* dB = dA + TBM * LDS_STR;
#pragma unroll
    for (int i = 0; i < 2; i++) {
      *(uint4*)(dA + ldRow[i] * LDS_STR + ldCol[i]) = ra[i];
      *(uint4*)(dB + ldRow[i] * LDS_STR + ldCol[i]) = rb[i];
    }
  };
  stage(0);
  commit(0);
  for (int kt = 0; kt < KT - 1; kt++) {    // KT >= 4 always
    __syncthreads();
    stage((kt + 1) * TBK);
    compute(kt & 1);
    commit((kt + 1) & 1);
  }
  __syncthreads();
  compute((KT - 1) & 1);
#endif

  // ---- epilogue: C/D layout lane = (m/8)*16 + n ; VGPR r -> m%8 ----
  const int cn_l  = lane & 15;
  const int cm_hi = (lane >> 4) << 3;
#pragma unroll
  for (int mt = 0; mt < 4; mt++) {
#pragma unroll
    for (int nt = 0; nt < 2; nt++) {
#pragma unroll
      for (int r = 0; r < 8; r++) {
        int gm = m0 + wm * 64 + mt * 16 + cm_hi + r;     // always < M (M % 128 == 0)
        int gn = n0 + wn * 32 + nt * 16 + cn_l;
        if (NGUARD && gn >= N) continue;
        float v = acc[mt][nt][r];
        if (MODE == 0) {
          C0[(size_t)gm * N + gn] = v;
        } else if (MODE == 1) {           // xz: even col -> x_raw, odd -> z
          int d = gn >> 1;
          float* dst = (gn & 1) ? C1 : C0;
          dst[(size_t)gm * (N >> 1) + d] = v;
        } else if (MODE == 2) {           // x_dbl split
          if (gn < DTR)                Cb[(size_t)gm * DTR + gn]            = f2bf(v);
          else if (gn < DTR + DSTATE)  C1[(size_t)gm * DSTATE + (gn - DTR)] = v;
          else                         C2[(size_t)gm * DSTATE + (gn - DTR - DSTATE)] = v;
        } else {                          // MODE 3: softplus(acc + b_dt[d])
          float t = v + bias[gn];
          float sp = (t > 20.0f) ? t : __logf(1.0f + __expf(t));
          C0[(size_t)gm * N + gn] = sp;
        }
      }
    }
  }
}

// ---------------- depthwise causal conv(4) + bias + SiLU ----------------
__global__ __launch_bounds__(256)
void conv_silu_kernel(const float* __restrict__ xraw, const float* __restrict__ cw,
                      const float* __restrict__ cb, float* __restrict__ xo,
                      __bf16* __restrict__ xob)
{
  constexpr int BPB = DI / 256;                       // blocks per batch
  int b = blockIdx.x / BPB;
  int d = (blockIdx.x % BPB) * 256 + threadIdx.x;
  float w0 = cw[d * 4 + 0], w1 = cw[d * 4 + 1], w2 = cw[d * 4 + 2], w3 = cw[d * 4 + 3];
  float bias = cb[d];
  const float* src  = xraw + (size_t)b * LSEQ * DI + d;
  float*       dst  = xo   + (size_t)b * LSEQ * DI + d;
  __bf16*      dstb = xob  + (size_t)b * LSEQ * DI + d;
  float h0 = 0.f, h1 = 0.f, h2 = 0.f;                 // x[l-3], x[l-2], x[l-1]
  for (int l = 0; l < LSEQ; l++) {
    float c = src[(size_t)l * DI];
    float a = h0 * w0 + h1 * w1 + h2 * w2 + c * w3 + bias;
    float s = silu_f(a);
    dst[(size_t)l * DI]  = s;
    dstb[(size_t)l * DI] = f2bf(s);
    h0 = h1; h1 = h2; h2 = c;
  }
}

// ---------------- selective scan (sequential in L, parallel over b*d) ----------------
__global__ __launch_bounds__(256)
void scan_kernel(const float* __restrict__ dtp, const float* __restrict__ xp,
                 const float* __restrict__ zp,  const float* __restrict__ Bp,
                 const float* __restrict__ Cp,  const float* __restrict__ Alog,
                 const float* __restrict__ Dp,  __bf16* __restrict__ ybf)
{
  constexpr int BPB = DI / 256;                       // blocks per batch
  int b = blockIdx.x / BPB;                           // wave-uniform
  int d = (blockIdx.x % BPB) * 256 + threadIdx.x;

  float An[DSTATE];
#pragma unroll
  for (int n = 0; n < DSTATE; n++) An[n] = -__expf(Alog[(size_t)d * DSTATE + n]);
  float Dd = Dp[d];
  float st[DSTATE];
#pragma unroll
  for (int n = 0; n < DSTATE; n++) st[n] = 0.f;

  for (int l = 0; l < LSEQ; l++) {
    size_t t    = (size_t)b * LSEQ + l;               // uniform -> scalar B/C loads
    size_t base = t * DI + d;
    float dtv = dtp[base];
    float xv  = xp[base];
    float zv  = zp[base];
    float4 Bq[4], Cq[4];
#pragma unroll
    for (int q = 0; q < 4; q++) {
      Bq[q] = *(const float4*)(Bp + t * DSTATE + q * 4);
      Cq[q] = *(const float4*)(Cp + t * DSTATE + q * 4);
    }
    const float* Bv = (const float*)Bq;
    const float* Cv = (const float*)Cq;
    float bx = dtv * xv;
    float y = 0.f;
#pragma unroll
    for (int n = 0; n < DSTATE; n++) {
      float dA = __expf(dtv * An[n]);
      st[n] = st[n] * dA + bx * Bv[n];
      y += st[n] * Cv[n];
    }
    y += Dd * xv;
    ybf[base] = f2bf(y * silu_f(zv));
  }
}

// ---------------- host launcher ----------------
extern "C" void kernel_launch(void* const* d_in, const int* in_sizes, int n_in,
                              void* d_out, int out_size, void* d_ws, size_t ws_size,
                              hipStream_t stream) {
  const float* hs     = (const float*)d_in[0];   // (2,2048,2048)
  const float* W_in   = (const float*)d_in[1];   // (8192,2048)
  const float* conv_w = (const float*)d_in[2];   // (4096,4)
  const float* conv_b = (const float*)d_in[3];   // (4096,)
  const float* W_x    = (const float*)d_in[4];   // (160,4096)
  const float* W_dt   = (const float*)d_in[5];   // (4096,128)
  const float* b_dt   = (const float*)d_in[6];   // (4096,)
  const float* A_log  = (const float*)d_in[7];   // (4096,16)
  const float* Dvec   = (const float*)d_in[8];   // (4096,)
  const float* W_out  = (const float*)d_in[9];   // (2048,4096)
  float* out = (float*)d_out;                    // (2,2048,2048)

  // ---- carve workspace (256B aligned slabs) ----
  char* ws = (char*)d_ws;
  size_t off = 0;
  auto carve = [&](size_t bytes) -> void* {
    void* p = ws + off;
    off += (bytes + 255) & ~(size_t)255;
    return p;
  };
  __bf16* hs_bf   = (__bf16*)carve((size_t)NTOK * DM * 2);
  __bf16* Win_bf  = (__bf16*)carve((size_t)EOUT * DM * 2);
  __bf16* Wx_bf   = (__bf16*)carve((size_t)XDBL * DI * 2);
  __bf16* Wdt_bf  = (__bf16*)carve((size_t)DI * DTR * 2);
  __bf16* Wout_bf = (__bf16*)carve((size_t)DM * DI * 2);
  float*  xraw    = (float*) carve((size_t)NTOK * DI * 4);
  float*  zbuf    = (float*) carve((size_t)NTOK * DI * 4);
  float*  xf      = (float*) carve((size_t)NTOK * DI * 4);
  __bf16* xbf     = (__bf16*)carve((size_t)NTOK * DI * 2);
  __bf16* dtlr    = (__bf16*)carve((size_t)NTOK * DTR * 2);
  float*  Bmat    = (float*) carve((size_t)NTOK * DSTATE * 4);
  float*  Cmat    = (float*) carve((size_t)NTOK * DSTATE * 4);
  float*  dtf     = (float*) carve((size_t)NTOK * DI * 4);
  __bf16* ybf     = (__bf16*)carve((size_t)NTOK * DI * 2);

  // ---- 1. weight/activation conversion to bf16 ----
  cvt_f32_bf16<<<2048, 256, 0, stream>>>(hs,    hs_bf,   (long)NTOK * DM);
  cvt_f32_bf16<<<2048, 256, 0, stream>>>(W_in,  Win_bf,  (long)EOUT * DM);
  cvt_f32_bf16<<<512,  256, 0, stream>>>(W_x,   Wx_bf,   (long)XDBL * DI);
  cvt_f32_bf16<<<512,  256, 0, stream>>>(W_dt,  Wdt_bf,  (long)DI * DTR);
  cvt_f32_bf16<<<2048, 256, 0, stream>>>(W_out, Wout_bf, (long)DM * DI);

  // ---- 2. xz GEMM (M=4096, N=8192, K=2048): deinterleave -> x_raw, z ----
  gemm_bf16_wmma<1, false><<<dim3(EOUT / TBN, NTOK / TBM), 256, 0, stream>>>(
      hs_bf, Win_bf, NTOK, EOUT, DM, xraw, zbuf, nullptr, nullptr, nullptr);

  // ---- 3. depthwise conv + SiLU ----
  conv_silu_kernel<<<(NBATCH * DI) / 256, 256, 0, stream>>>(xraw, conv_w, conv_b, xf, xbf);

  // ---- 4. x_dbl GEMM (M=4096, N=160, K=4096): split dt_lr / B / C ----
  gemm_bf16_wmma<2, true><<<dim3((XDBL + TBN - 1) / TBN, NTOK / TBM), 256, 0, stream>>>(
      xbf, Wx_bf, NTOK, XDBL, DI, nullptr, Bmat, Cmat, dtlr, nullptr);

  // ---- 5. dt GEMM (M=4096, N=4096, K=128) + bias + softplus ----
  gemm_bf16_wmma<3, false><<<dim3(DI / TBN, NTOK / TBM), 256, 0, stream>>>(
      dtlr, Wdt_bf, NTOK, DI, DTR, dtf, nullptr, nullptr, nullptr, b_dt);

  // ---- 6. selective scan, fused D*x and silu(z) gate, emits bf16 y ----
  scan_kernel<<<(NBATCH * DI) / 256, 256, 0, stream>>>(
      dtf, xf, zbuf, Bmat, Cmat, A_log, Dvec, ybf);

  // ---- 7. output GEMM (M=4096, N=2048, K=4096) -> d_out f32 ----
  gemm_bf16_wmma<0, false><<<dim3(DM / TBN, NTOK / TBM), 256, 0, stream>>>(
      ybf, Wout_bf, NTOK, DM, DI, out, nullptr, nullptr, nullptr, nullptr);
}
